// GraphAutoencoder_80874234184212
// MI455X (gfx1250) — matrix operand order repaired and checked by reading, not compiled, not run on hardware
//
#include <hip/hip_runtime.h>
#include <hip/hip_bf16.h>

typedef __attribute__((ext_vector_type(16))) _Float16 v16h;
typedef __attribute__((ext_vector_type(8)))  _Float16 v8h;
typedef __attribute__((ext_vector_type(8)))  float    v8f;

// ---------------------------------------------------------------------------
// Operand packing: f32 -> f16 with zero padding so GEMM fragment loads are
// unconditional 128-bit vector loads.
//   Ah [Mp, Kp]  row-major   (Mp = ceil16(M), Kp = ceil32(K))
//   WT [Ncp, Kp] row-major = W^T with padding (Ncp = ceil16(Nc))
// ---------------------------------------------------------------------------
__global__ void pack_a_kernel(const float* __restrict__ src, _Float16* __restrict__ dst,
                              int m, int k, int mp, int kp) {
  const size_t tot = (size_t)mp * kp;
  for (size_t idx = (size_t)blockIdx.x * blockDim.x + threadIdx.x; idx < tot;
       idx += (size_t)gridDim.x * blockDim.x) {
    const int row = (int)(idx / kp), col = (int)(idx % kp);
    dst[idx] = (row < m && col < k) ? (_Float16)src[(size_t)row * k + col] : (_Float16)0.f;
  }
}

__global__ void pack_wt_kernel(const float* __restrict__ w, _Float16* __restrict__ dst,
                               int k, int nc, int ncp, int kp) {
  const size_t tot = (size_t)ncp * kp;
  for (size_t idx = (size_t)blockIdx.x * blockDim.x + threadIdx.x; idx < tot;
       idx += (size_t)gridDim.x * blockDim.x) {
    const int n = (int)(idx / kp), kk = (int)(idx % kp);
    dst[idx] = (n < nc && kk < k) ? (_Float16)w[(size_t)kk * nc + n] : (_Float16)0.f;
  }
}

// ---------------------------------------------------------------------------
// WMMA GEMM: C[M,Nc] = Ah[M,Kp] @ WT[Nc,Kp]^T  (f16 operands, f32 accum).
// One wave per 16x16 output tile. Branch-free inner loop:
//   4 x global_load_b128  +  1 x v_wmma_f32_16x16x32_f16  per 32-K step.
// CDNA5 fragment layouts (CDNA5 ISA §7.12.2), hi = lane>=16:
//   A: lane&15 = row M; halves 0-7 = K[k0+8hi ..], halves 8-15 = K[k0+16+8hi ..]
//   B: lane&15 = col N; halves 0-7 = K[k0+8hi ..], halves 8-15 = K[k0+16+8hi ..]
//   C/D: VGPR r = row r + 8hi, lane&15 = col
// ---------------------------------------------------------------------------
__global__ void wmma_gemm_kernel(const _Float16* __restrict__ Ah,
                                 const _Float16* __restrict__ WT,
                                 float* __restrict__ C,
                                 int M, int Kp, int Nc, int tilesM, int tilesN) {
  const int lane = threadIdx.x & 31;
  const int wave = blockIdx.x * (blockDim.x >> 5) + (threadIdx.x >> 5);
  const int tm = wave / tilesN;
  const int tn = wave % tilesN;
  if (tm >= tilesM) return;                 // wave-uniform: EXEC stays all-1s

  const int hi    = lane >> 4;
  const int mrow  = tm * 16 + (lane & 15);
  const int ncolf = tn * 16 + (lane & 15);

  const _Float16* arow = Ah + (size_t)mrow * Kp + hi * 8;    // 16B aligned
  const _Float16* brow = WT + (size_t)ncolf * Kp + hi * 8;   // 16B aligned

  v8f acc = {};
  for (int k0 = 0; k0 < Kp; k0 += 32) {
    union { v16h v; v8h h[2]; } a, b;
    a.h[0] = *(const v8h*)(arow + k0);        // K = k0+8hi   .. +7
    a.h[1] = *(const v8h*)(arow + k0 + 16);   // K = k0+16+8hi.. +7
    b.h[0] = *(const v8h*)(brow + k0);        // K = k0+8hi   .. +7
    b.h[1] = *(const v8h*)(brow + k0 + 16);   // K = k0+16+8hi.. +7
    acc = __builtin_amdgcn_wmma_f32_16x16x32_f16(false, a.v, false, b.v,
                                                 (short)0, acc, false, false);
  }

  const int rbase = tm * 16 + hi * 8;
#pragma unroll
  for (int r = 0; r < 8; ++r) {
    const int row = rbase + r;
    if (row < M && ncolf < Nc) C[(size_t)row * Nc + ncolf] = acc[r];
  }
}

// ---------------------------------------------------------------------------
// Attention scores per node: es[i] = h[i].a_src, ed[i] = h[i].a_dst
// ---------------------------------------------------------------------------
__global__ void node_dots_kernel(const float* __restrict__ h,
                                 const float* __restrict__ asrc,
                                 const float* __restrict__ adst,
                                 float* __restrict__ es, float* __restrict__ ed,
                                 int n, int d) {
  const int i = blockIdx.x * blockDim.x + threadIdx.x;
  if (i >= n) return;
  const float* row = h + (size_t)i * d;
  float s = 0.f, t = 0.f;
  for (int j = 0; j < d; ++j) { const float v = row[j]; s += v * asrc[j]; t += v * adst[j]; }
  es[i] = s; ed[i] = t;
}

__global__ void init_node_kernel(float* __restrict__ m, float* __restrict__ ss, int n) {
  const int i = blockIdx.x * blockDim.x + threadIdx.x;
  if (i < n) { m[i] = -__builtin_inff(); ss[i] = 0.f; }
}

__global__ void init_out_bias_kernel(float* __restrict__ out,
                                     const float* __restrict__ bias,
                                     int n, int d) {
  const size_t tot = (size_t)n * d;
  for (size_t idx = (size_t)blockIdx.x * blockDim.x + threadIdx.x; idx < tot;
       idx += (size_t)gridDim.x * blockDim.x)
    out[idx] = bias[idx % d];
}

__device__ __forceinline__ void atomic_max_f32(float* addr, float v) {
  if (v >= 0.f) atomicMax((int*)addr, __float_as_int(v));
  else          atomicMin((unsigned int*)addr, __float_as_uint(v));
}

__device__ __forceinline__ void edge_sr(const int* __restrict__ ei, int E, int idx,
                                        int& s, int& r) {
  if (idx < E) { s = ei[idx]; r = ei[E + idx]; }  // edge_index rows 0 / 1
  else         { s = idx - E; r = s; }            // appended self loops
}

__device__ __forceinline__ float leaky(float v) { return v > 0.f ? v : 0.2f * v; }

__global__ void edge_max_kernel(const int* __restrict__ ei,
                                const float* __restrict__ es,
                                const float* __restrict__ ed,
                                float* __restrict__ m, int E, int total) {
  const int idx = blockIdx.x * blockDim.x + threadIdx.x;
  if (idx >= total) return;
  int s, r; edge_sr(ei, E, idx, s, r);
  atomic_max_f32(&m[r], leaky(es[s] + ed[r]));
}

__global__ void edge_exp_kernel(const int* __restrict__ ei,
                                const float* __restrict__ es,
                                const float* __restrict__ ed,
                                const float* __restrict__ m,
                                float* __restrict__ ep, float* __restrict__ ss,
                                int E, int total) {
  const int idx = blockIdx.x * blockDim.x + threadIdx.x;
  if (idx >= total) return;
  int s, r; edge_sr(ei, E, idx, s, r);
  const float p = __expf(leaky(es[s] + ed[r]) - m[r]);
  ep[idx] = p;
  atomicAdd(&ss[r], p);
}

__global__ void edge_alpha_kernel(const int* __restrict__ ei,
                                  float* __restrict__ ep,
                                  const float* __restrict__ ss,
                                  int E, int total) {
  const int idx = blockIdx.x * blockDim.x + threadIdx.x;
  if (idx >= total) return;
  int s, r; edge_sr(ei, E, idx, s, r);
  ep[idx] = ep[idx] / ss[r];
}

// out[recv] += alpha * h[send]; grid-stride over (E+N)*d scalar adds.
// h / out both fit in MI455X's 192MB L2, so this is L2 traffic, not HBM.
__global__ void edge_aggr_kernel(const int* __restrict__ ei,
                                 const float* __restrict__ alpha,
                                 const float* __restrict__ h,
                                 float* __restrict__ out,
                                 int E, int total, int d) {
  const unsigned long long nwork = (unsigned long long)total * d;
  for (unsigned long long g = (unsigned long long)blockIdx.x * blockDim.x + threadIdx.x;
       g < nwork; g += (unsigned long long)gridDim.x * blockDim.x) {
    const int idx = (int)(g / d);
    const int j   = (int)(g % d);
    int s, r; edge_sr(ei, E, idx, s, r);
    if (j == 0) __builtin_prefetch(&h[(size_t)s * d], 0, 1);  // global_prefetch_b8
    atomicAdd(&out[(size_t)r * d + j], alpha[idx] * h[(size_t)s * d + j]);
  }
}

__global__ void relu_kernel(float* __restrict__ x, size_t n) {
  for (size_t i = (size_t)blockIdx.x * blockDim.x + threadIdx.x; i < n;
       i += (size_t)gridDim.x * blockDim.x)
    x[i] = fmaxf(x[i], 0.f);
}

__global__ void init_pool_kernel(float* __restrict__ pooled, float* __restrict__ cnt,
                                 int B, int d) {
  const int i = blockIdx.x * blockDim.x + threadIdx.x;
  if (i < B * d) pooled[i] = 0.f;
  if (i < B) cnt[i] = 0.f;
}

__global__ void pool_kernel(const float* __restrict__ latent,
                            const int* __restrict__ batch,
                            float* __restrict__ pooled, float* __restrict__ cnt,
                            int n, int d) {
  const int i = blockIdx.x * blockDim.x + threadIdx.x;
  if (i >= n) return;
  const int b = batch[i];
  atomicAdd(&cnt[b], 1.f);
  const float* row = latent + (size_t)i * d;
  for (int j = 0; j < d; ++j) atomicAdd(&pooled[(size_t)b * d + j], row[j]);
}

__global__ void classify_kernel(const float* __restrict__ pooled,
                                const float* __restrict__ cnt,
                                const float* __restrict__ Wc1, const float* __restrict__ bc1,
                                const float* __restrict__ Wc2, const float* __restrict__ bc2,
                                float* __restrict__ noise,
                                int B, int dlat, int dh, int ncls) {
  const int b = blockIdx.x * blockDim.x + threadIdx.x;
  if (b >= B) return;
  const float inv = 1.f / fmaxf(cnt[b], 1.f);
  float z[64];
  for (int j = 0; j < dh; ++j) {
    float s = bc1[j];
    for (int k = 0; k < dlat; ++k) s += (pooled[(size_t)b * dlat + k] * inv) * Wc1[(size_t)k * dh + j];
    z[j] = fmaxf(s, 0.f);
  }
  for (int c = 0; c < ncls; ++c) {
    float s = bc2[c];
    for (int j = 0; j < dh; ++j) s += z[j] * Wc2[(size_t)j * ncls + c];
    noise[(size_t)b * ncls + c] = s;
  }
}

// ---------------------------------------------------------------------------
// Host-side per-layer driver (all launches on `stream`; graph-capture safe)
// ---------------------------------------------------------------------------
static void gat_layer(const float* xin, const float* W, const float* as_,
                      const float* ad_, const float* bias,
                      _Float16* ah, _Float16* wt,
                      float* hbuf, float* outbuf,
                      float* es, float* ed, float* mb, float* ss, float* ealpha,
                      const int* ei, int N, int E, int din, int dout, bool relu,
                      hipStream_t stream) {
  const int total = E + N;
  const int Mp  = ((N    + 15) / 16) * 16;
  const int Kp  = ((din  + 31) / 32) * 32;
  const int Ncp = ((dout + 15) / 16) * 16;
  const int tilesM = Mp / 16, tilesN = Ncp / 16;

  // pack operands to padded f16
  {
    const size_t tot = (size_t)Mp * Kp;
    unsigned long long blk = (tot + 255) / 256; if (blk > 262144ull) blk = 262144ull;
    pack_a_kernel<<<(int)blk, 256, 0, stream>>>(xin, ah, N, din, Mp, Kp);
    const size_t totw = (size_t)Ncp * Kp;
    pack_wt_kernel<<<(int)((totw + 255) / 256), 256, 0, stream>>>(W, wt, din, dout, Ncp, Kp);
  }
  // h = x @ W  (WMMA)
  {
    const int waves = tilesM * tilesN;
    wmma_gemm_kernel<<<(waves + 3) / 4, 128, 0, stream>>>(ah, wt, hbuf, N, Kp, dout,
                                                          tilesM, tilesN);
  }
  node_dots_kernel<<<(N + 255) / 256, 256, 0, stream>>>(hbuf, as_, ad_, es, ed, N, dout);
  init_node_kernel<<<(N + 255) / 256, 256, 0, stream>>>(mb, ss, N);
  {
    const size_t tot = (size_t)N * dout;
    init_out_bias_kernel<<<(int)((tot + 255) / 256), 256, 0, stream>>>(outbuf, bias, N, dout);
  }
  edge_max_kernel  <<<(total + 255) / 256, 256, 0, stream>>>(ei, es, ed, mb, E, total);
  edge_exp_kernel  <<<(total + 255) / 256, 256, 0, stream>>>(ei, es, ed, mb, ealpha, ss, E, total);
  edge_alpha_kernel<<<(total + 255) / 256, 256, 0, stream>>>(ei, ealpha, ss, E, total);
  {
    const unsigned long long nwork = (unsigned long long)total * dout;
    unsigned long long blk = (nwork + 255) / 256;
    if (blk > 262144ull) blk = 262144ull;
    edge_aggr_kernel<<<(int)blk, 256, 0, stream>>>(ei, ealpha, hbuf, outbuf, E, total, dout);
  }
  if (relu) {
    const size_t tot = (size_t)N * dout;
    unsigned long long blk = (tot + 255) / 256;
    if (blk > 262144ull) blk = 262144ull;
    relu_kernel<<<(int)blk, 256, 0, stream>>>(outbuf, tot);
  }
}

extern "C" void kernel_launch(void* const* d_in, const int* in_sizes, int n_in,
                              void* d_out, int out_size, void* d_ws, size_t ws_size,
                              hipStream_t stream) {
  // ---- inputs (setup_inputs order) ----
  const float* x     = (const float*)d_in[0];
  const int*   ei    = (const int*)  d_in[1];
  const int*   batch = (const int*)  d_in[2];
  const float* W1 = (const float*)d_in[3],  *a1s = (const float*)d_in[4],
             * a1d = (const float*)d_in[5],  *b1  = (const float*)d_in[6];
  const float* W2 = (const float*)d_in[7],  *a2s = (const float*)d_in[8],
             * a2d = (const float*)d_in[9],  *b2  = (const float*)d_in[10];
  const float* W3 = (const float*)d_in[11], *a3s = (const float*)d_in[12],
             * a3d = (const float*)d_in[13], *b3  = (const float*)d_in[14];
  const float* W4 = (const float*)d_in[15], *a4s = (const float*)d_in[16],
             * a4d = (const float*)d_in[17], *b4  = (const float*)d_in[18];
  const float* Wc1 = (const float*)d_in[19], *bc1 = (const float*)d_in[20];
  const float* Wc2 = (const float*)d_in[21], *bc2 = (const float*)d_in[22];

  // ---- dims ----
  const int N    = in_sizes[2];            // batch has one entry per node
  const int E    = in_sizes[1] / 2;        // edge_index is [2, E]
  const int DIN  = in_sizes[0] / N;        // 3
  const int DHID = in_sizes[6];            // 128
  const int DLAT = in_sizes[10];           // 64
  const int DC   = in_sizes[20];           // 32
  const int NCLS = in_sizes[22];           // 2
  const int B    = (out_size - N * DIN - N * DLAT) / NCLS;
  const int Mp   = ((N + 15) / 16) * 16;

  // ---- output regions (reference return order, flattened) ----
  float* recon  = (float*)d_out;                         // [N, DIN]
  float* latent = recon + (size_t)N * DIN;               // [N, DLAT]
  float* noise  = latent + (size_t)N * DLAT;             // [B, NCLS]

  // ---- workspace carve-up (offsets kept 64B-aligned for b128 loads) ----
  float* ws = (float*)d_ws;
  size_t o = 0;
  auto align16 = [&]() { o = (o + 15) & ~(size_t)15; };
  float* hbuf   = ws + o; o += (size_t)N * DHID;         // transformed features
  float* aggbuf = ws + o; o += (size_t)N * DHID;         // layer-1/3 outputs
  align16();
  _Float16* ah  = (_Float16*)(ws + o); o += (size_t)Mp * DHID / 2;  // packed A (f16)
  align16();
  _Float16* wt  = (_Float16*)(ws + o); o += (size_t)DHID * DHID / 2;// packed W^T (f16)
  align16();
  float* es     = ws + o; o += N;
  float* ed     = ws + o; o += N;
  float* mb     = ws + o; o += N;
  float* ss     = ws + o; o += N;
  float* ealpha = ws + o; o += (size_t)(E + N);
  float* pooled = ws + o; o += (size_t)B * DLAT;
  float* cntb   = ws + o; o += B;

  // ---- encoder ----
  gat_layer(x,      W1, a1s, a1d, b1, ah, wt, hbuf, aggbuf, es, ed, mb, ss, ealpha,
            ei, N, E, DIN,  DHID, /*relu=*/true,  stream);
  gat_layer(aggbuf, W2, a2s, a2d, b2, ah, wt, hbuf, latent, es, ed, mb, ss, ealpha,
            ei, N, E, DHID, DLAT, /*relu=*/false, stream);
  // ---- decoder ----
  gat_layer(latent, W3, a3s, a3d, b3, ah, wt, hbuf, aggbuf, es, ed, mb, ss, ealpha,
            ei, N, E, DLAT, DHID, /*relu=*/true,  stream);
  gat_layer(aggbuf, W4, a4s, a4d, b4, ah, wt, hbuf, recon,  es, ed, mb, ss, ealpha,
            ei, N, E, DHID, DIN,  /*relu=*/false, stream);

  // ---- per-graph mean pooling + classifier ----
  init_pool_kernel<<<(B * DLAT + 255) / 256, 256, 0, stream>>>(pooled, cntb, B, DLAT);
  pool_kernel<<<(N + 255) / 256, 256, 0, stream>>>(latent, batch, pooled, cntb, N, DLAT);
  classify_kernel<<<(B + 63) / 64, 64, 0, stream>>>(pooled, cntb, Wc1, bc1, Wc2, bc2,
                                                    noise, B, DLAT, DC, NCLS);
}
